// PriorNetwork_24172075942128
// MI455X (gfx1250) — compile-verified, matrix operand order checked by reading, not run on hardware
//
#include <hip/hip_runtime.h>

// ---------------------------------------------------------------------------
// PriorNetwork on MI455X (gfx1250, wave32, WMMA):
//   1-NN argmin over s[b,n] = t2[n] - 2*dot(codes[b], train[n])  (q2 constant)
//   then mu/logstd = relu(train[argmin] @ W1 + b1) @ {Wu,Ws} + {bu,bs}
//
// Distance GEMM: 26.2 GFLOP over ~26MB unique bytes (fits L2) -> compute
// bound. Use split-f16 (x = hi + lo) with 3x v_wmma_f32_16x16x32_f16 per
// K=32 chunk for ~fp32 accuracy at f16 matrix rates. Tiles are streamed
// into LDS with GLOBAL_LOAD_ASYNC_TO_LDS_B128 (ASYNCcnt) double-buffered
// against the WMMA consumer. MLP is tiny -> exact v_wmma_f32_16x16x4_f32.
// ---------------------------------------------------------------------------

typedef __attribute__((ext_vector_type(16))) _Float16 v16h;
typedef __attribute__((ext_vector_type(8)))  float    v8f;
typedef __attribute__((ext_vector_type(2)))  float    v2f;

constexpr int D    = 64;    // code dim
constexpr int H    = 512;   // hidden dim
constexpr int BM   = 128;   // batch rows per WG in argmin kernel (8 waves x 16)
constexpr int NSEG = 64;    // N-dimension segments for 2-stage argmin

// workgroup-relative LDS byte offset of a __shared__ object
__device__ __forceinline__ unsigned lds_offset(const void* p) {
  return (unsigned)(size_t)(__attribute__((address_space(3))) const char*)p;
}

// async 16B global -> LDS copy (VFLAT GV mode), tracked by ASYNCcnt
__device__ __forceinline__ void async_copy_b128(unsigned lds_byte_off, const void* gsrc) {
  asm volatile("global_load_async_to_lds_b128 %0, %1, off"
               :: "v"(lds_byte_off), "v"(gsrc) : "memory");
}
__device__ __forceinline__ void wait_asynccnt0() {
  asm volatile("s_wait_asynccnt 0x0" ::: "memory");
}

// ---------------- kernel: split f32 -> (f16 hi, f16 lo) --------------------
__global__ void k_split(const float* __restrict__ src, _Float16* __restrict__ hi,
                        _Float16* __restrict__ lo, long n) {
  long i = (long)blockIdx.x * blockDim.x + threadIdx.x;
  if (i >= n) return;
  float x = src[i];
  _Float16 h = (_Float16)x;
  hi[i] = h;
  lo[i] = (_Float16)(x - (float)h);
}

// ---------------- kernel: row squared norms t2[n] --------------------------
__global__ void k_rownorm(const float* __restrict__ t, float* __restrict__ t2, int n) {
  int r = blockIdx.x * blockDim.x + threadIdx.x;
  if (r >= n) return;
  const float4* p = (const float4*)(t + (size_t)r * D);
  float s = 0.f;
#pragma unroll
  for (int i = 0; i < D / 4; ++i) {
    float4 v = p[i];
    s += v.x * v.x + v.y * v.y + v.z * v.z + v.w * v.w;
  }
  t2[r] = s;
}

// ---------------- kernel: fused GEMM + partial argmin ----------------------
constexpr int SH_STRIDE = 80;   // halves per LDS row (pad 64 -> 80: no bank conflicts, 16B aligned)

__global__ void __launch_bounds__(256)
k_argmin(const _Float16* __restrict__ cH, const _Float16* __restrict__ cL,
         const _Float16* __restrict__ tH, const _Float16* __restrict__ tL,
         const float* __restrict__ t2,
         float* __restrict__ partV, int* __restrict__ partI,
         int Ntiles, int Nrows) {
  __shared__ _Float16 shH[2][16 * SH_STRIDE];
  __shared__ _Float16 shL[2][16 * SH_STRIDE];
  __shared__ float    sh2[2][16];

  const int t    = threadIdx.x;
  const int wave = t >> 5;
  const int lane = t & 31;
  const int lo16 = lane & 15;
  const int hi16 = lane >> 4;

  const int seg     = blockIdx.x;
  const int rowBase = blockIdx.y * BM + wave * 16;

  // A fragments (codes rows, split f16), K=64 -> two K-chunks of 32.
  // 16-bit A layout: lane<16 holds K {0..7,16..23}, lane>=16 holds {8..15,24..31}.
  v16h aH[2], aL[2];
#pragma unroll
  for (int kc = 0; kc < 2; ++kc) {
    const _Float16* pH = cH + (size_t)(rowBase + lo16) * D + kc * 32 + 8 * hi16;
    const _Float16* pL = cL + (size_t)(rowBase + lo16) * D + kc * 32 + 8 * hi16;
#pragma unroll
    for (int j = 0; j < 8; ++j) {
      aH[kc][j] = pH[j];  aH[kc][8 + j] = pH[16 + j];
      aL[kc][j] = pL[j];  aL[kc][8 + j] = pL[16 + j];
    }
  }

  const int per = (Ntiles + NSEG - 1) / NSEG;
  const int t0  = seg * per;
  const int t1  = min(t0 + per, Ntiles);

  float mv[8];
  int   mi[8];
#pragma unroll
  for (int r = 0; r < 8; ++r) { mv[r] = 3.4e38f; mi[r] = 0; }

  // Cooperative async tile loader: each 16x64-half tile is 2KB = 128 x 16B.
  // Threads 0..127 stream the hi tile, 128..255 the lo tile (wave-uniform
  // split, EXEC stays full). 16B aligned on both global and LDS sides.
  auto load_tile = [&](int nbase, int buf) {
    const int ti   = t & 127;
    const int row  = ti >> 3;          // 8 chunks of 16B per row
    const int k    = (ti & 7) * 8;     // half-index within row
    const size_t g = (size_t)(nbase + row) * D + k;
    if (t < 128) {
      async_copy_b128(lds_offset(&shH[buf][row * SH_STRIDE + k]), tH + g);
      __builtin_prefetch(tH + g + 32 * D, 0, 1);   // two tiles ahead
    } else {
      async_copy_b128(lds_offset(&shL[buf][row * SH_STRIDE + k]), tL + g);
      __builtin_prefetch(tL + g + 32 * D, 0, 1);
    }
    if (t < 16) sh2[buf][t] = t2[nbase + t];
  };

  if (t0 < t1) load_tile(t0 * 16, 0);
  wait_asynccnt0();
  __syncthreads();

  for (int tt = t0; tt < t1; ++tt) {
    const int buf = (tt - t0) & 1;
    if (tt + 1 < t1) load_tile((tt + 1) * 16, buf ^ 1);   // async into other buffer

    v8f acc = {};
#pragma unroll
    for (int kc = 0; kc < 2; ++kc) {
      v16h bH, bL;
      const _Float16* qH = &shH[buf][lo16 * SH_STRIDE + kc * 32 + 8 * hi16];
      const _Float16* qL = &shL[buf][lo16 * SH_STRIDE + kc * 32 + 8 * hi16];
#pragma unroll
      for (int j = 0; j < 8; ++j) {
        bH[j] = qH[j];  bH[8 + j] = qH[16 + j];
        bL[j] = qL[j];  bL[8 + j] = qL[16 + j];
      }
      // dot = hiA*hiB + hiA*loB + loA*hiB  (~fp32 accuracy), f32 accumulate
      acc = __builtin_amdgcn_wmma_f32_16x16x32_f16(false, aH[kc], false, bH, (short)0, acc, false, false);
      acc = __builtin_amdgcn_wmma_f32_16x16x32_f16(false, aH[kc], false, bL, (short)0, acc, false, false);
      acc = __builtin_amdgcn_wmma_f32_16x16x32_f16(false, aL[kc], false, bH, (short)0, acc, false, false);
    }

    const float t2v = sh2[buf][lo16];
    const int   n   = tt * 16 + lo16;
#pragma unroll
    for (int r = 0; r < 8; ++r) {
      float s = fmaf(-2.f, acc[r], t2v);
      if (s < mv[r]) { mv[r] = s; mi[r] = n; }   // ties keep older (smaller) index
    }
    wait_asynccnt0();      // next tile's LDS writes have landed
    __syncthreads();
  }

  // C layout: acc[r] <-> row (rowBase + r + 8*(lane>=16)), col n%16 == lane%16.
  // Reduce (min,idx) across the 16 lanes of each half-wave.
#pragma unroll
  for (int off = 1; off < 16; off <<= 1) {
#pragma unroll
    for (int r = 0; r < 8; ++r) {
      float ov = __shfl_xor(mv[r], off, 32);
      int   oi = __shfl_xor(mi[r], off, 32);
      if (ov < mv[r] || (ov == mv[r] && oi < mi[r])) { mv[r] = ov; mi[r] = oi; }
    }
  }
  if (lo16 == 0) {
#pragma unroll
    for (int r = 0; r < 8; ++r) {
      const int b = rowBase + r + 8 * hi16;
      partV[(size_t)b * NSEG + seg] = mv[r];
      partI[(size_t)b * NSEG + seg] = mi[r];
    }
  }
}

// ---------------- kernel: reduce partial argmins ---------------------------
__global__ void k_reduce(const float* __restrict__ partV, const int* __restrict__ partI,
                         int* __restrict__ chosen, int Bsz) {
  int b = blockIdx.x * blockDim.x + threadIdx.x;
  if (b >= Bsz) return;
  float mv = 3.4e38f;
  int   mi = 0x7fffffff;
  for (int s = 0; s < NSEG; ++s) {
    float v = partV[(size_t)b * NSEG + s];
    int   i = partI[(size_t)b * NSEG + s];
    if (v < mv || (v == mv && i < mi)) { mv = v; mi = i; }
  }
  chosen[b] = mi;
}

// ---------------- kernel: gather + MLP (f32 WMMA 16x16x4) ------------------
constexpr int PREV_S = 72;    // padded strides (bank-conflict free, 8B aligned)
constexpr int H1_S   = 520;

__global__ void __launch_bounds__(256)
k_mlp(const float* __restrict__ train, const int* __restrict__ chosen,
      const float* __restrict__ W1, const float* __restrict__ b1,
      const float* __restrict__ Wu, const float* __restrict__ bu,
      const float* __restrict__ Ws, const float* __restrict__ bs,
      float* __restrict__ out, int Bsz) {
  __shared__ float prev[16 * PREV_S];
  __shared__ float h1[16 * H1_S];

  const int t       = threadIdx.x;
  const int rowBase = blockIdx.x * 16;

  for (int i = t; i < 16 * D; i += 256) {
    int row = i >> 6, k = i & 63;
    prev[row * PREV_S + k] = train[(size_t)chosen[rowBase + row] * D + k];
  }
  __syncthreads();

  const int wave = t >> 5, lane = t & 31, lo16 = lane & 15, hi16 = lane >> 4;

  // GEMM1: h1 = relu(prev @ W1 + b1); wave w owns H columns [w*64, w*64+64).
  // f32 16x16x4 A layout: lane holds A[lane%16][4*kk + 2*(lane>=16) + {0,1}].
  for (int ct = 0; ct < 4; ++ct) {
    const int col = wave * 64 + ct * 16 + lo16;
    v8f acc = {};
#pragma unroll
    for (int kk = 0; kk < D / 4; ++kk) {
      const int k = kk * 4 + 2 * hi16;
      v2f a; a[0] = prev[lo16 * PREV_S + k];        a[1] = prev[lo16 * PREV_S + k + 1];
      v2f w; w[0] = W1[(size_t)k * H + col];        w[1] = W1[(size_t)(k + 1) * H + col];
      acc = __builtin_amdgcn_wmma_f32_16x16x4_f32(false, a, false, w, (short)0, acc, false, false);
    }
    const float bias = b1[col];
#pragma unroll
    for (int r = 0; r < 8; ++r) {
      float v = acc[r] + bias;
      h1[(r + 8 * hi16) * H1_S + col] = v > 0.f ? v : 0.f;
    }
  }
  __syncthreads();

  // GEMM2: waves 0-3 -> mu columns, waves 4-7 -> logstd columns (K = 512).
  const float* W  = (wave < 4) ? Wu : Ws;
  const float* bb = (wave < 4) ? bu : bs;
  const int col = (wave & 3) * 16 + lo16;
  v8f acc = {};
  for (int kk = 0; kk < H / 4; ++kk) {
    const int k = kk * 4 + 2 * hi16;
    v2f a; a[0] = h1[lo16 * H1_S + k];       a[1] = h1[lo16 * H1_S + k + 1];
    v2f w; w[0] = W[(size_t)k * D + col];    w[1] = W[(size_t)(k + 1) * D + col];
    acc = __builtin_amdgcn_wmma_f32_16x16x4_f32(false, a, false, w, (short)0, acc, false, false);
  }
  const float  bias  = bb[col];
  const size_t obase = (wave < 4) ? 0 : (size_t)Bsz * D;
#pragma unroll
  for (int r = 0; r < 8; ++r) {
    const int b = rowBase + r + 8 * hi16;
    out[obase + (size_t)b * D + col] = acc[r] + bias;
  }
}

// ---------------------------------------------------------------------------
extern "C" void kernel_launch(void* const* d_in, const int* in_sizes, int n_in,
                              void* d_out, int out_size, void* d_ws, size_t ws_size,
                              hipStream_t stream) {
  const float* codes = (const float*)d_in[0];
  const float* train = (const float*)d_in[1];
  const float* W1    = (const float*)d_in[2];
  const float* b1    = (const float*)d_in[3];
  const float* Wu    = (const float*)d_in[4];
  const float* bu    = (const float*)d_in[5];
  const float* Ws    = (const float*)d_in[6];
  const float* bs    = (const float*)d_in[7];

  const int Bsz = in_sizes[0] / D;   // 2048
  const int N   = in_sizes[1] / D;   // 100000

  // workspace carve-out (256B aligned)
  char*  ws  = (char*)d_ws;
  size_t off = 0;
  auto carve = [&](size_t bytes) -> char* {
    char* p = ws + off;
    off = (off + bytes + 255) & ~(size_t)255;
    return p;
  };
  float*    t2v    = (float*)   carve((size_t)N * 4);
  _Float16* tH     = (_Float16*)carve((size_t)N * D * 2);
  _Float16* tL     = (_Float16*)carve((size_t)N * D * 2);
  _Float16* cH     = (_Float16*)carve((size_t)Bsz * D * 2);
  _Float16* cL     = (_Float16*)carve((size_t)Bsz * D * 2);
  float*    partV  = (float*)   carve((size_t)Bsz * NSEG * 4);
  int*      partI  = (int*)     carve((size_t)Bsz * NSEG * 4);
  int*      chosen = (int*)     carve((size_t)Bsz * 4);
  (void)ws_size; (void)n_in; (void)out_size;

  const long totTrain = (long)N * D;
  const long totCodes = (long)Bsz * D;

  k_split  <<<dim3((unsigned)((totTrain + 255) / 256)), 256, 0, stream>>>(train, tH, tL, totTrain);
  k_split  <<<dim3((unsigned)((totCodes + 255) / 256)), 256, 0, stream>>>(codes, cH, cL, totCodes);
  k_rownorm<<<dim3((unsigned)((N + 255) / 256)),        256, 0, stream>>>(train, t2v, N);

  const int Ntiles = N / 16;
  k_argmin<<<dim3(NSEG, Bsz / BM), 256, 0, stream>>>(cH, cL, tH, tL, t2v, partV, partI, Ntiles, N);
  k_reduce<<<dim3((unsigned)((Bsz + 255) / 256)), 256, 0, stream>>>(partV, partI, chosen, Bsz);

  k_mlp<<<dim3((unsigned)(Bsz / 16)), 256, 0, stream>>>(train, chosen, W1, b1, Wu, bu, Ws, bs,
                                                        (float*)d_out, Bsz);
}
